// DCTWindowAttention_85813446574228
// MI455X (gfx1250) — matrix operand order repaired and checked by reading, not compiled, not run on hardware
//
#include <hip/hip_runtime.h>

// Fused Swin window attention for gfx1250 (MI455X).
// f16 WMMA (v_wmma_f32_16x16x32_f16) for all GEMMs, f32 accumulation,
// in-register softmax via wave32 shfl_xor, all intermediates in LDS,
// double-buffered B fragments + global_prefetch_b8 for the next x block.

typedef _Float16 half_t;
typedef __attribute__((ext_vector_type(16))) _Float16 v16h;
typedef __attribute__((ext_vector_type(8)))  float    v8f;

#define THREADS 256
#define NWAVES  8
#define WSTR    130                       // padded row stride (halves) for 128-wide mats
#define VSTR    18                        // transposed-V row stride (halves)
#define WQ_H    (384 * WSTR)
#define WP_H    (128 * WSTR)
#define XB_H    (NWAVES * 16 * WSTR)
#define WAVE_H  (16*WSTR /*q*/ + 16*WSTR /*k*/ + 128*VSTR /*vT*/ + 16*32 /*p*/ + 16*WSTR /*o*/)
#define LDS_BYTES ((WQ_H + WP_H + XB_H) * 2 + (4*256 + 384 + 128) * 4 + NWAVES * WAVE_H * 2)

static __device__ __forceinline__ v16h load_a16(const half_t* p) {
  // A 16x32 f16 fragment: 16 contiguous K values of one row
  v16h a;
#pragma unroll
  for (int i = 0; i < 8; ++i)  a[i] = p[i];
#pragma unroll
  for (int i = 8; i < 16; ++i) a[i] = p[i];
  return a;
}
static __device__ __forceinline__ v16h load_b16(const half_t* p) {
  // B 32x16 f16 fragment: 16 contiguous K values of one column
  v16h b;
#pragma unroll
  for (int i = 0; i < 16; ++i) b[i] = p[i];
  return b;
}
static __device__ __forceinline__ void load_b4(const half_t* pb, v16h* b) {
  b[0] = load_b16(pb);
  b[1] = load_b16(pb + 32);
  b[2] = load_b16(pb + 64);
  b[3] = load_b16(pb + 96);
}
// One 16x16 tile, K=128, two interleaved accumulators (chain depth 2).
static __device__ __forceinline__ v8f gemm4(const v16h* a, const v16h* b) {
  v8f acc0 = {}, acc1 = {};
  acc0 = __builtin_amdgcn_wmma_f32_16x16x32_f16(false, a[0], false, b[0], (short)0, acc0, false, false);
  acc1 = __builtin_amdgcn_wmma_f32_16x16x32_f16(false, a[1], false, b[1], (short)0, acc1, false, false);
  acc0 = __builtin_amdgcn_wmma_f32_16x16x32_f16(false, a[2], false, b[2], (short)0, acc0, false, false);
  acc1 = __builtin_amdgcn_wmma_f32_16x16x32_f16(false, a[3], false, b[3], (short)0, acc1, false, false);
  v8f r;
#pragma unroll
  for (int i = 0; i < 8; ++i) r[i] = acc0[i] + acc1[i];
  return r;
}

__global__ void __launch_bounds__(THREADS)
swin_window_attention(const float* __restrict__ x,
                      const float* __restrict__ qkv_w,
                      const float* __restrict__ qkv_b,
                      const float* __restrict__ proj_w,
                      const float* __restrict__ proj_b,
                      const float* __restrict__ bias_table,
                      const int*   __restrict__ rel_pos_index,
                      float* __restrict__ out,
                      int n_groups) {
  extern __shared__ char lds_raw[];
  half_t* wq = (half_t*)lds_raw;            // qkv_w f16, stride WSTR
  half_t* wp = wq + WQ_H;                   // proj_w f16
  half_t* xb = wp + WP_H;                   // staged x, token-major, 8 windows
  float*  bm = (float*)(xb + XB_H);         // bias[h][i][j], 4x16x16
  float*  bq = bm + 4 * 256;                // qkv_b
  float*  bp = bq + 384;                    // proj_b
  half_t* wvbase = (half_t*)(bp + 128);     // per-wave scratch

  const int tid  = threadIdx.x;
  const int wave = tid >> 5;
  const int lane = tid & 31;
  const int lo16 = lane & 15;
  const int hi   = lane >> 4;               // 0 or 1
  const int mb   = hi << 3;                 // D-frag row base for this lane half

  half_t* sq  = wvbase + wave * WAVE_H;     // Q  16x128 (stride WSTR)
  half_t* sk  = sq + 16 * WSTR;             // K  16x128
  half_t* svT = sk + 16 * WSTR;             // V^T 128x16 (stride VSTR)
  half_t* sp  = svT + 128 * VSTR;           // P  16x32 (K-padded with zeros)
  half_t* so  = sp + 16 * 32;               // attn out 16x128
  half_t* sres = sq;                        // proj result reuses Q region
  const half_t* zero16 = sp + 16;           // 16 halves guaranteed zero (K-pad row 0)

  // ---- one-time staging: weights -> f16 LDS, bias matrix ----
  for (int i = tid; i < 384 * 128; i += THREADS)
    wq[(i >> 7) * WSTR + (i & 127)] = (half_t)qkv_w[i];
  for (int i = tid; i < 128 * 128; i += THREADS)
    wp[(i >> 7) * WSTR + (i & 127)] = (half_t)proj_w[i];
  for (int i = tid; i < 4 * 256; i += THREADS)
    bm[i] = bias_table[rel_pos_index[i & 255] * 4 + (i >> 8)];
  for (int i = tid; i < 384; i += THREADS) bq[i] = qkv_b[i];
  for (int i = tid; i < 128; i += THREADS) bp[i] = proj_b[i];
  for (int i = lane; i < 16 * 32; i += 32) sp[i] = (half_t)0;  // zero K-pad region

  const float scale = 0.17677669529663687f;  // 1/sqrt(32)

  for (int g = blockIdx.x; g < n_groups; g += gridDim.x) {
    const int gw = g & 3;                    // 8-window group along W
    const int hw = (g >> 2) & 31;            // window row
    const int b  = g >> 7;                   // batch
    const int h0 = hw * 4;
    const int w0 = gw * 32;

    __syncthreads();
    // ---- stage x: NCHW global -> token-major f16 LDS (coalesced along W) ----
    for (int i = tid; i < 128 * 4 * 32; i += THREADS) {
      int col = i & 31, r = (i >> 5) & 3, c = i >> 7;
      float v = x[(((size_t)b * 128 + c) * 128 + (h0 + r)) * 128 + (w0 + col)];
      int win = col >> 2, tok = r * 4 + (col & 3);
      xb[(win * 16 + tok) * WSTR + c] = (half_t)v;
    }
    __syncthreads();

    // ---- prefetch next group's x block into cache while we compute ----
    {
      const int gn = g + gridDim.x;
      if (gn < n_groups) {
        const int gw2 = gn & 3, hw2 = (gn >> 2) & 31, b2 = gn >> 7;
        const int h02 = hw2 * 4, w02 = gw2 * 32;
        // 128 ch x 4 rows, each row-strip of 32 floats = one 128B line
        for (int i = tid; i < 512; i += THREADS) {
          int c = i >> 2, r = i & 3;
          __builtin_prefetch(
              &x[(((size_t)b2 * 128 + c) * 128 + (h02 + r)) * 128 + w02], 0, 3);
        }
      }
    }

    // ================= per-wave: one window =================
    const half_t* xw = xb + wave * 16 * WSTR;
    v16h ax[4];
    {
      const half_t* pr = xw + lo16 * WSTR + (hi << 4);
#pragma unroll
      for (int kc = 0; kc < 4; ++kc) ax[kc] = load_a16(pr + kc * 32);
    }

    // ---- QKV GEMM: (16x128) x (128x384), three passes, B double-buffered ----
    const half_t* wq_base = wq + lo16 * WSTR + (hi << 4);
    v16h bfr[2][4];
    load_b4(wq_base, bfr[0]);
#pragma unroll
    for (int nt = 0; nt < 8; ++nt) {               // Q tiles (pre-scaled)
      if (nt < 7) load_b4(wq_base + (nt + 1) * 16 * WSTR, bfr[(nt + 1) & 1]);
      else        load_b4(wq_base + 128 * WSTR, bfr[(nt + 1) & 1]);  // first K tile
      v8f acc = gemm4(ax, bfr[nt & 1]);
      const float bias = bq[(nt << 4) + lo16];
      half_t* dst = sq + (nt << 4) + lo16;
#pragma unroll
      for (int r = 0; r < 8; ++r)
        dst[(mb + r) * WSTR] = (half_t)((acc[r] + bias) * scale);
    }
#pragma unroll
    for (int nt = 0; nt < 8; ++nt) {               // K tiles
      if (nt < 7) load_b4(wq_base + (128 + (nt + 1) * 16) * WSTR, bfr[(nt + 1) & 1]);
      else        load_b4(wq_base + 256 * WSTR, bfr[(nt + 1) & 1]);  // first V tile
      v8f acc = gemm4(ax, bfr[nt & 1]);
      const float bias = bq[128 + (nt << 4) + lo16];
      half_t* dst = sk + (nt << 4) + lo16;
#pragma unroll
      for (int r = 0; r < 8; ++r)
        dst[(mb + r) * WSTR] = (half_t)(acc[r] + bias);
    }
#pragma unroll
    for (int nt = 0; nt < 8; ++nt) {               // V tiles, stored transposed
      if (nt < 7) load_b4(wq_base + (256 + (nt + 1) * 16) * WSTR, bfr[(nt + 1) & 1]);
      v8f acc = gemm4(ax, bfr[nt & 1]);
      const float bias = bq[256 + (nt << 4) + lo16];
      half_t* dst = svT + ((nt << 4) + lo16) * VSTR + mb;
#pragma unroll
      for (int r = 0; r < 8; ++r)
        dst[r] = (half_t)(acc[r] + bias);          // contiguous -> one ds_store_b128
    }

    // ---- attention per head: QK^T (K=32) -> softmax -> P V (K padded to 32) ----
#pragma unroll
    for (int h = 0; h < 4; ++h) {
      v16h aq = load_a16(sq + lo16 * WSTR + h * 32 + (hi << 4));
      v16h bk = load_b16(sk + lo16 * WSTR + h * 32 + (hi << 4));
      v8f s = {};
      s = __builtin_amdgcn_wmma_f32_16x16x32_f16(false, aq, false, bk,
                                                 (short)0, s, false, false);
      const float* bmh = bm + h * 256;
      float p8[8];
#pragma unroll
      for (int r = 0; r < 8; ++r) {
        float a = s[r] + bmh[(mb + r) * 16 + lo16];
        float m = a;
#pragma unroll
        for (int off = 1; off < 16; off <<= 1)
          m = fmaxf(m, __shfl_xor(m, off, 32));   // row max within 16-lane half
        float e = __expf(a - m);
        float sum = e;
#pragma unroll
        for (int off = 1; off < 16; off <<= 1)
          sum += __shfl_xor(sum, off, 32);        // row sum
        p8[r] = e / sum;
      }
#pragma unroll
      for (int r = 0; r < 8; ++r)
        sp[(mb + r) * 32 + lo16] = (half_t)p8[r];

      v16h ap = load_a16(sp + lo16 * 32 + (hi << 4));  // K 16..31 read as zeros
#pragma unroll
      for (int dt = 0; dt < 2; ++dt) {
        // Low lanes read a V^T row (16 contiguous halves); high lanes (K rows
        // 16..31) read a guaranteed-zero LDS region instead of cndmasks.
        const half_t* pv = hi ? zero16
                              : svT + (h * 32 + dt * 16 + lo16) * VSTR;
        v16h bv = load_b16(pv);
        v8f o = {};
        o = __builtin_amdgcn_wmma_f32_16x16x32_f16(false, ap, false, bv,
                                                   (short)0, o, false, false);
        half_t* dst = so + h * 32 + dt * 16 + lo16;
#pragma unroll
        for (int r = 0; r < 8; ++r)
          dst[(mb + r) * WSTR] = (half_t)o[r];
      }
    }

    // ---- output projection: (16x128) x (128x128), B double-buffered ----
    v16h ao[4];
    {
      const half_t* pr = so + lo16 * WSTR + (hi << 4);
#pragma unroll
      for (int kc = 0; kc < 4; ++kc) ao[kc] = load_a16(pr + kc * 32);
    }
    const half_t* wp_base = wp + lo16 * WSTR + (hi << 4);
    load_b4(wp_base, bfr[0]);
#pragma unroll
    for (int nt = 0; nt < 8; ++nt) {
      if (nt < 7) load_b4(wp_base + (nt + 1) * 16 * WSTR, bfr[(nt + 1) & 1]);
      v8f acc = gemm4(ao, bfr[nt & 1]);
      const float bias = bp[(nt << 4) + lo16];
      half_t* dst = sres + (nt << 4) + lo16;
#pragma unroll
      for (int r = 0; r < 8; ++r)
        dst[(mb + r) * WSTR] = (half_t)(acc[r] + bias);
    }

    __syncthreads();
    // ---- restriped coalesced store back to NCHW ----
    for (int i = tid; i < 128 * 4 * 32; i += THREADS) {
      int col = i & 31, r = (i >> 5) & 3, c = i >> 7;
      int win = col >> 2, tok = r * 4 + (col & 3);
      const half_t* src = wvbase + win * WAVE_H;   // that wave's result block
      out[(((size_t)b * 128 + c) * 128 + (h0 + r)) * 128 + (w0 + col)] =
          (float)src[tok * WSTR + c];
    }
  }
}

extern "C" void kernel_launch(void* const* d_in, const int* in_sizes, int n_in,
                              void* d_out, int out_size, void* d_ws, size_t ws_size,
                              hipStream_t stream) {
  (void)in_sizes; (void)n_in; (void)out_size; (void)d_ws; (void)ws_size;
  const float* x          = (const float*)d_in[0];
  const float* qkv_w      = (const float*)d_in[1];
  const float* qkv_b      = (const float*)d_in[2];
  const float* proj_w     = (const float*)d_in[3];
  const float* proj_b     = (const float*)d_in[4];
  const float* bias_table = (const float*)d_in[5];
  const int*   rpi        = (const int*)d_in[6];
  float*       out        = (float*)d_out;

  const int n_groups = 16 * 32 * 4;  // B * nHw * (nWw/8) = 2048
  hipFuncSetAttribute((const void*)swin_window_attention,
                      hipFuncAttributeMaxDynamicSharedMemorySize, (int)LDS_BYTES);
  swin_window_attention<<<512, THREADS, LDS_BYTES, stream>>>(
      x, qkv_w, qkv_b, proj_w, proj_b, bias_table, rpi, out, n_groups);
}
